// SetAbstraction_52003464020759
// MI455X (gfx1250) — matrix operand order repaired and checked by reading, not compiled, not run on hardware
//
#include <hip/hip_runtime.h>
#include <hip/hip_bf16.h>

// ---------------------------------------------------------------------------
// PointNet++ Set Abstraction for MI455X (gfx1250, wave32, WMMA).
//
// Roofline: MLP = 17.4 GFLOP f16-WMMA (trivial); intermediates (134-268MB)
// are L2-resident (192MB). Real cost is the serial FPS scan (2048 steps) and
// the top-32 selection; both are register/LDS-resident designs below.
// ---------------------------------------------------------------------------

#define NB     8
#define NPTS   8192
#define NS     2048
#define KNB    32
#define CIN    64
#define BSK    (NB * NS * KNB)      // 524288 grouped points
#define R2     0.01f                // RADIUS^2
#define EPS_BN 1e-5f

typedef __attribute__((ext_vector_type(16))) _Float16 v16h;
typedef __attribute__((ext_vector_type(8)))  float    v8f;
typedef __attribute__((ext_vector_type(4)))  float    v4f;

// ---------------------------------------------------------------------------
// Prep: convert weights to f16 (pad layer0 67 -> 96), zero BN stat buffers.
// ---------------------------------------------------------------------------
__global__ __launch_bounds__(256) void prep_kernel(
    const float* __restrict__ W0, const float* __restrict__ W1,
    const float* __restrict__ W2,
    _Float16* __restrict__ W0h, _Float16* __restrict__ W1h,
    _Float16* __restrict__ W2h, float* __restrict__ stats) {
  int t = blockIdx.x * 256 + threadIdx.x;
  if (t < 3 * 256) stats[t] = 0.0f;
  if (t < 64 * 96) {
    int o = t / 96, c = t % 96;
    W0h[t] = (_Float16)(c < 67 ? W0[o * 67 + c] : 0.0f);
  }
  if (t < 64 * 64)  W1h[t] = (_Float16)W1[t];
  if (t < 128 * 64) W2h[t] = (_Float16)W2[t];
}

// ---------------------------------------------------------------------------
// FPS: one workgroup (1024 thr = 32 waves) per batch. Points + running
// min-distances live in registers; argmax via wave32 shfl_xor reductions.
// Writes new_xyz directly into d_out (first B*S*3 floats).
// ---------------------------------------------------------------------------
__global__ __launch_bounds__(1024) void fps_kernel(
    const float* __restrict__ xyz, float* __restrict__ newxyz) {
  const int b = blockIdx.x, tid = threadIdx.x;
  const float* base = xyz + (size_t)b * NPTS * 3;

  float px[8], py[8], pz[8], dd[8];
#pragma unroll
  for (int j = 0; j < 8; ++j) {
    int i = j * 1024 + tid;
    px[j] = base[i * 3 + 0];
    py[j] = base[i * 3 + 1];
    pz[j] = base[i * 3 + 2];
    dd[j] = 1e10f;
  }

  __shared__ float s_c[3];
  __shared__ int   s_cur;
  __shared__ float wb[32];
  __shared__ int   wbi[32];
  if (tid == 0) { s_c[0] = base[0]; s_c[1] = base[1]; s_c[2] = base[2]; }
  __syncthreads();

  const int wave = tid >> 5, lane = tid & 31;
  for (int s = 0; s < NS; ++s) {
    float cx = s_c[0], cy = s_c[1], cz = s_c[2];
    if (tid == 0) {                       // cents[s] coords -> new_xyz[b,s]
      float* o = newxyz + ((size_t)b * NS + s) * 3;
      o[0] = cx; o[1] = cy; o[2] = cz;
    }
    float best = -1.0f; int bi = 0;
#pragma unroll
    for (int j = 0; j < 8; ++j) {
      float dx = px[j] - cx, dy = py[j] - cy, dz = pz[j] - cz;
      float d = dx * dx + dy * dy + dz * dz;
      d = fminf(dd[j], d);
      dd[j] = d;
      int i = j * 1024 + tid;
      if (d > best || (d == best && i < bi)) { best = d; bi = i; }
    }
#pragma unroll
    for (int off = 16; off > 0; off >>= 1) {
      float od = __shfl_xor(best, off, 32);
      int   oi = __shfl_xor(bi,   off, 32);
      if (od > best || (od == best && oi < bi)) { best = od; bi = oi; }
    }
    if (lane == 0) { wb[wave] = best; wbi[wave] = bi; }
    __syncthreads();
    if (wave == 0) {
      float vb = wb[lane]; int vi = wbi[lane];
#pragma unroll
      for (int off = 16; off > 0; off >>= 1) {
        float od = __shfl_xor(vb, off, 32);
        int   oi = __shfl_xor(vi, off, 32);
        if (od > vb || (od == vb && oi < vi)) { vb = od; vi = oi; }
      }
      if (lane == 0) s_cur = vi;
    }
    __syncthreads();
    int cur = s_cur;
    if ((cur & 1023) == tid) {            // owner publishes new centroid coords
      int j = cur >> 10;
      float ox = px[0], oy = py[0], oz = pz[0];
#pragma unroll
      for (int jj = 1; jj < 8; ++jj)
        if (j == jj) { ox = px[jj]; oy = py[jj]; oz = pz[jj]; }
      s_c[0] = ox; s_c[1] = oy; s_c[2] = oz;
    }
    __syncthreads();
  }
}

// ---------------------------------------------------------------------------
// KNN: thread-per-centroid, register-resident sorted top-32, xyz streamed in
// 24KB LDS chunks (all-lanes-same-index reads = LDS broadcast).
// ---------------------------------------------------------------------------
#define CHUNK 2048
__global__ __launch_bounds__(256) void knn_kernel(
    const float* __restrict__ xyz, const float* __restrict__ newxyz,
    int* __restrict__ idxout) {
  __shared__ float s3[CHUNK * 3];
  const int b = blockIdx.y;
  const int s = blockIdx.x * 256 + threadIdx.x;
  const float* base = xyz + (size_t)b * NPTS * 3;
  const float* c = newxyz + ((size_t)b * NS + s) * 3;
  const float cx = c[0], cy = c[1], cz = c[2];

  float dist[KNB]; int idx[KNB];
#pragma unroll
  for (int k = 0; k < KNB; ++k) { dist[k] = 3.4e38f; idx[k] = 0; }

  for (int c0 = 0; c0 < NPTS; c0 += CHUNK) {
    __syncthreads();
    for (int t = threadIdx.x; t < CHUNK * 3; t += 256)
      s3[t] = base[(size_t)c0 * 3 + t];
    __syncthreads();
    for (int i = 0; i < CHUNK; ++i) {
      float dx = s3[i * 3 + 0] - cx;
      float dy = s3[i * 3 + 1] - cy;
      float dz = s3[i * 3 + 2] - cz;
      float d = dx * dx + dy * dy + dz * dz;
      if (d < dist[KNB - 1]) {            // branch-light sorted insert
        int ii = c0 + i;
#pragma unroll
        for (int j = KNB - 1; j >= 1; --j) {
          if (dist[j] > d) {
            bool sh = dist[j - 1] > d;
            dist[j] = sh ? dist[j - 1] : d;
            idx[j]  = sh ? idx[j - 1]  : ii;
          }
        }
        if (dist[0] > d) { dist[0] = d; idx[0] = ii; }
      }
    }
  }
  int i0 = idx[0];
  int* o = idxout + ((size_t)b * NS + s) * KNB;
#pragma unroll
  for (int k = 0; k < KNB; ++k) o[k] = (dist[k] > R2) ? i0 : idx[k];
}

// ---------------------------------------------------------------------------
// Gather: build X0 f32 [BSK][96] = [xyz[idx]-center(3) | points[idx](64) | 0pad].
// ---------------------------------------------------------------------------
__global__ __launch_bounds__(256) void gather_kernel(
    const float* __restrict__ xyz, const float* __restrict__ points,
    const float* __restrict__ newxyz, const int* __restrict__ idx,
    float* __restrict__ X0) {
  int p  = blockIdx.x * 256 + threadIdx.x;   // grouped point id
  int bs = p / KNB;
  int b  = bs / NS;
  int i  = idx[p];
  const float* c = newxyz + (size_t)bs * 3;
  const float* q = xyz + ((size_t)b * NPTS + i) * 3;
  float* x = X0 + (size_t)p * 96;
  x[0] = q[0] - c[0]; x[1] = q[1] - c[1]; x[2] = q[2] - c[2];
  const float* pt = points + ((size_t)b * NPTS + i) * CIN;
#pragma unroll
  for (int j = 0; j < CIN; ++j) x[3 + j] = pt[j];
#pragma unroll
  for (int j = 67; j < 96; ++j) x[j] = 0.0f;
}

// ---------------------------------------------------------------------------
// WMMA GEMM + fused BN-stat epilogue. One wave per 16-row M tile.
//   Y[m][n] = sum_k act(A[m][k]) * W[n][k] + bias[n]
// act = identity (layer0) or relu(a*scale[k]+shift[k]) (fused BN of prev layer).
// A frag: row M=lane&15, 16 contiguous k at kb=kt*32+(lane<16?0:16).
// B frag: weight row n=nt*16+(lane&15), 16 halfs at kt*32+(lane<16?0:16).
// D frag: lane's 8 accs all share channel n -> cheap per-channel sum/sumsq.
// ---------------------------------------------------------------------------
template <int NT, int KT, bool FUSE>
__global__ __launch_bounds__(256) void gemm_bn_kernel(
    const float* __restrict__ A, const _Float16* __restrict__ Bw,
    const float* __restrict__ bias, const float* __restrict__ scale,
    const float* __restrict__ shift, float* __restrict__ Y,
    float* __restrict__ gstats) {
  constexpr int KK = KT * 32;
  constexpr int NN = NT * 16;
  __shared__ float s_sum[NN], s_sq[NN];
  __shared__ float s_as[KK], s_ab[KK];

  const int tid = threadIdx.x;
  for (int i = tid; i < NN; i += 256) { s_sum[i] = 0.0f; s_sq[i] = 0.0f; }
  if constexpr (FUSE) {
    for (int i = tid; i < KK; i += 256) { s_as[i] = scale[i]; s_ab[i] = shift[i]; }
  }
  __syncthreads();

  const int wave = tid >> 5, lane = tid & 31;
  const int mtile = blockIdx.x * 8 + wave;
  const int row = mtile * 16 + (lane & 15);
  const float* arow = A + (size_t)row * KK;
  const int loHalf = (lane < 16);

  v8f zero = {};
  v8f acc[NT];
#pragma unroll
  for (int nt = 0; nt < NT; ++nt) acc[nt] = zero;

#pragma unroll
  for (int kt = 0; kt < KT; ++kt) {
    const int kb = kt * 32 + (loHalf ? 0 : 16);
    v4f t0 = *(const v4f*)(arow + kb);
    v4f t1 = *(const v4f*)(arow + kb + 4);
    v4f t2 = *(const v4f*)(arow + kb + 8);
    v4f t3 = *(const v4f*)(arow + kb + 12);
    float fv[16];
#pragma unroll
    for (int e = 0; e < 4; ++e) {
      fv[e] = t0[e]; fv[4 + e] = t1[e]; fv[8 + e] = t2[e]; fv[12 + e] = t3[e];
    }
    v16h a;
#pragma unroll
    for (int e = 0; e < 16; ++e) {
      float x = fv[e];
      if constexpr (FUSE) {
        int kc = kb + e;
        x = x * s_as[kc] + s_ab[kc];
        x = x > 0.0f ? x : 0.0f;
      }
      a[e] = (_Float16)x;
    }
#pragma unroll
    for (int nt = 0; nt < NT; ++nt) {
      const int col = nt * 16 + (lane & 15);
      v16h bfrag = *(const v16h*)(Bw + (size_t)col * KK + kt * 32 + (loHalf ? 0 : 16));
      acc[nt] = __builtin_amdgcn_wmma_f32_16x16x32_f16(
          false, a, false, bfrag, (short)0, acc[nt], false, false);
    }
  }

#pragma unroll
  for (int nt = 0; nt < NT; ++nt) {
    const int n = nt * 16 + (lane & 15);
    const float bv = bias[n];
    float ls = 0.0f, lq = 0.0f;
#pragma unroll
    for (int v = 0; v < 8; ++v) {
      float y = acc[nt][v] + bv;
      int r = mtile * 16 + v + (loHalf ? 0 : 8);
      Y[(size_t)r * NN + n] = y;
      ls += y; lq += y * y;
    }
    atomicAdd(&s_sum[n], ls);
    atomicAdd(&s_sq[n], lq);
  }
  __syncthreads();
  for (int i = tid; i < NN; i += 256) {
    atomicAdd(&gstats[i], s_sum[i]);
    atomicAdd(&gstats[NN + i], s_sq[i]);
  }
}

// ---------------------------------------------------------------------------
// BN finalize: fold mean/var/gamma/beta into per-channel scale/shift.
// ---------------------------------------------------------------------------
__global__ void finalize_bn_kernel(const float* __restrict__ gstats,
                                   const float* __restrict__ g,
                                   const float* __restrict__ beta,
                                   float* __restrict__ scale,
                                   float* __restrict__ shift,
                                   int n_ch, float invcnt) {
  int n = threadIdx.x;
  if (n < n_ch) {
    float m  = gstats[n] * invcnt;
    float v  = gstats[n_ch + n] * invcnt - m * m;
    float iv = rsqrtf(v + EPS_BN);
    float sc = iv * g[n];
    scale[n] = sc;
    shift[n] = beta[n] - m * sc;
  }
}

// ---------------------------------------------------------------------------
// Final: relu(BN(Y2)) then max over K neighbors -> out[B,S,128].
// ---------------------------------------------------------------------------
__global__ __launch_bounds__(128) void maxpool_kernel(
    const float* __restrict__ Y2, const float* __restrict__ scale,
    const float* __restrict__ shift, float* __restrict__ outfeat) {
  const int bs = blockIdx.x;
  const int n  = threadIdx.x;
  const float sc = scale[n], sh = shift[n];
  const float* y = Y2 + (size_t)bs * KNB * 128 + n;
  float m = 0.0f;                       // relu >= 0
  for (int k = 0; k < KNB; ++k) {
    float v = y[(size_t)k * 128] * sc + sh;
    v = v > 0.0f ? v : 0.0f;
    m = v > m ? v : m;
  }
  outfeat[(size_t)bs * 128 + n] = m;
}

// ---------------------------------------------------------------------------
extern "C" void kernel_launch(void* const* d_in, const int* in_sizes, int n_in,
                              void* d_out, int out_size, void* d_ws, size_t ws_size,
                              hipStream_t stream) {
  const float* xyz    = (const float*)d_in[0];
  const float* points = (const float*)d_in[1];
  const float* W0 = (const float*)d_in[2];
  const float* b0 = (const float*)d_in[3];
  const float* g0 = (const float*)d_in[4];
  const float* be0 = (const float*)d_in[5];
  const float* W1 = (const float*)d_in[6];
  const float* b1 = (const float*)d_in[7];
  const float* g1 = (const float*)d_in[8];
  const float* be1 = (const float*)d_in[9];
  const float* W2 = (const float*)d_in[10];
  const float* b2 = (const float*)d_in[11];
  const float* g2 = (const float*)d_in[12];
  const float* be2 = (const float*)d_in[13];

  float* out     = (float*)d_out;
  float* newxyz  = out;                              // [B,S,3]
  float* outfeat = out + (size_t)NB * NS * 3;        // [B,S,128]

  // --- workspace layout (256B aligned) ---
  size_t off = 0;
  auto take = [&](size_t bytes) {
    size_t o = off;
    off = (off + bytes + 255) & ~(size_t)255;
    return o;
  };
  size_t oW0h = take(64 * 96 * sizeof(_Float16));
  size_t oW1h = take(64 * 64 * sizeof(_Float16));
  size_t oW2h = take(128 * 64 * sizeof(_Float16));
  size_t oSt  = take(3 * 256 * sizeof(float));       // 3 layers x [sum|sumsq]
  size_t oAff = take(6 * 128 * sizeof(float));       // 3 layers x [scale|shift]
  size_t oIdx = take((size_t)BSK * sizeof(int));
  size_t oRA  = take((size_t)BSK * 96 * sizeof(float));   // X0, later Y1
  size_t oRB  = take((size_t)BSK * 128 * sizeof(float));  // Y0, later Y2
  if (ws_size < off) return;                          // deterministic guard

  char* ws = (char*)d_ws;
  _Float16* W0h = (_Float16*)(ws + oW0h);
  _Float16* W1h = (_Float16*)(ws + oW1h);
  _Float16* W2h = (_Float16*)(ws + oW2h);
  float* stats  = (float*)(ws + oSt);
  float* st0 = stats, *st1 = stats + 256, *st2 = stats + 512;
  float* aff = (float*)(ws + oAff);
  float* sc0 = aff,        *sh0 = aff + 128;
  float* sc1 = aff + 256,  *sh1 = aff + 384;
  float* sc2 = aff + 512,  *sh2 = aff + 640;
  int*   idx = (int*)(ws + oIdx);
  float* X0  = (float*)(ws + oRA);
  float* Y1  = (float*)(ws + oRA);                   // reuses X0 (dead by then)
  float* Y0  = (float*)(ws + oRB);
  float* Y2  = (float*)(ws + oRB);                   // reuses Y0 (dead by then)

  const float invcnt = 1.0f / (float)BSK;

  prep_kernel<<<32, 256, 0, stream>>>(W0, W1, W2, W0h, W1h, W2h, stats);
  fps_kernel<<<NB, 1024, 0, stream>>>(xyz, newxyz);
  knn_kernel<<<dim3(NS / 256, NB), 256, 0, stream>>>(xyz, newxyz, idx);
  gather_kernel<<<BSK / 256, 256, 0, stream>>>(xyz, points, newxyz, idx, X0);

  // layer0: [BSK,96] x W0h[64,96]^T -> Y0 [BSK,64]   (12 wmma / wave)
  gemm_bn_kernel<4, 3, false><<<BSK / 128, 256, 0, stream>>>(
      X0, W0h, b0, nullptr, nullptr, Y0, st0);
  finalize_bn_kernel<<<1, 128, 0, stream>>>(st0, g0, be0, sc0, sh0, 64, invcnt);

  // layer1: relu(BN(Y0)) x W1h[64,64]^T -> Y1 [BSK,64]   (8 wmma / wave)
  gemm_bn_kernel<4, 2, true><<<BSK / 128, 256, 0, stream>>>(
      Y0, W1h, b1, sc0, sh0, Y1, st1);
  finalize_bn_kernel<<<1, 128, 0, stream>>>(st1, g1, be1, sc1, sh1, 64, invcnt);

  // layer2: relu(BN(Y1)) x W2h[128,64]^T -> Y2 [BSK,128]   (16 wmma / wave)
  gemm_bn_kernel<8, 2, true><<<BSK / 128, 256, 0, stream>>>(
      Y1, W2h, b2, sc1, sh1, Y2, st2);
  finalize_bn_kernel<<<1, 128, 0, stream>>>(st2, g2, be2, sc2, sh2, 128, invcnt);

  maxpool_kernel<<<NB * NS, 128, 0, stream>>>(Y2, sc2, sh2, outfeat);
}
